// SNN_42185168781498
// MI455X (gfx1250) — matrix operand order repaired and checked by reading, not compiled, not run on hardware
//
#include <hip/hip_runtime.h>

// ---------------------------------------------------------------------------
// Recurrent spiking net (snntorch RLeaky x2), MI455X / gfx1250.
//
// Phase 1 (parallel, WMMA): cur1[n, j] = sum_c x[c, t, s] * W1[j, c] + b1[j]
//   where n = s*T + t  (the exact order the sequential phase consumes).
//   GEMM M=131072, N=256, K=64 -> v_wmma_f32_16x16x32_f16, K split 2x32.
//   x (32MB) is L2-resident (192MB L2) so layout gathers are cheap.
//
// Phase 2 (sequential, persistent 1-workgroup): 131072 LIF steps.
//   V1 (256x256 f32) staged in LDS with +1 row padding (bank-conflict-free),
//   spike set kept as a double-buffered 256-bit ballot mask -> exactly ONE
//   __syncthreads() per time step and no LDS atomics on the critical path.
// ---------------------------------------------------------------------------

typedef __attribute__((ext_vector_type(16))) _Float16 v16h;
typedef __attribute__((ext_vector_type(8)))  float    v8f;

#define C_   64
#define T_   128
#define S_   1024
#define H_   256
#define THRV 1.0f
#define BETA 0.9f

// ---------------------------------------------------------------------------
// Phase 1: WMMA GEMM producing input currents for one chunk of samples.
// Block = 256 threads (8 waves). Each wave computes two 16x16 D tiles along N
// (N = 256 = 16 tiles) for the block's 16-row slab of n.
// ---------------------------------------------------------------------------
__global__ __launch_bounds__(256) void snn_gemm_cur(
    const float* __restrict__ x,   // [C, T, S]
    const float* __restrict__ W1,  // [H, C]
    const float* __restrict__ b1,  // [H]
    float* __restrict__ cur,       // [schunk*T, H]
    int chunk, int schunk)
{
  const int wave = threadIdx.x >> 5;
  const int lane = threadIdx.x & 31;
  const int m    = lane & 15;   // A row / B col / D col within tile
  const int hv   = lane >> 4;   // lane half: selects K sub-range / D row half

  const int nbase = blockIdx.x * 16;        // local n of tile row 0
  const int nl    = nbase + m;              // this lane's A row (local n)
  const int t     = nl & (T_ - 1);
  const int s     = chunk * schunk + (nl >> 7);   // T_ == 128
  const float* xcol = x + (size_t)t * S_ + s;     // + c * (T_*S_)

  // ---- A tiles (16x32 f16, two K-chunks), per ISA 16-bit A layout:
  // lane<16 holds K in {0..7, 16..23}; lane>=16 holds K in {8..15, 24..31}.
  v16h a0, a1;
#pragma unroll
  for (int e = 0; e < 16; ++e) {
    const int cc = (e < 8) ? (hv * 8 + e) : (16 + hv * 8 + (e - 8));
    a0[e] = (_Float16)xcol[(size_t)cc        * (T_ * S_)];
    a1[e] = (_Float16)xcol[(size_t)(cc + 32) * (T_ * S_)];
  }

  // ---- Two N-tiles per wave: jt = 2*wave, 2*wave+1
#pragma unroll
  for (int q = 0; q < 2; ++q) {
    const int jt = wave * 2 + q;
    const int j  = jt * 16 + m;              // B's N index == D column
    const float* w1row = W1 + j * C_;

    // B tile 32x16 f16: lane<16 holds K {0..7,16..23}, lane>=16 K {8..15,24..31}, packed 2/VGPR.
    v16h b0, b1v;
#pragma unroll
    for (int e = 0; e < 16; ++e) {
      const int c0 = (e < 8) ? (hv * 8 + e) : (16 + hv * 8 + (e - 8));
      b0[e]  = (_Float16)w1row[c0];
      b1v[e] = (_Float16)w1row[c0 + 32];
    }

    v8f acc = {};
    acc = __builtin_amdgcn_wmma_f32_16x16x32_f16(false, a0, false, b0,
                                                 (short)0, acc, false, false);
    acc = __builtin_amdgcn_wmma_f32_16x16x32_f16(false, a1, false, b1v,
                                                 (short)0, acc, false, false);

    const float bias = b1[j];
    // D layout: VGPR r -> row r + 8*hv, column m.
#pragma unroll
    for (int r = 0; r < 8; ++r) {
      const int row = r + 8 * hv;
      cur[(size_t)(nbase + row) * H_ + j] = acc[r] + bias;
    }
  }
}

// ---------------------------------------------------------------------------
// Phase 2: persistent single-workgroup sequential recurrence over one chunk.
// 256 threads; thread j owns neuron j of layer 1. Dynamic LDS:
//   Vp   : 256 x 257 f32 (padded V1)  -> bank-conflict-free gathers
//   w2s  : 256 f32 (W2 row)
//   mask : 2 x 8 u32 ping-pong spike bitmask (one ballot word per wave)
// One barrier per step: iteration n writes mask[n&1], reads mask[1-(n&1)].
// ---------------------------------------------------------------------------
__global__ __launch_bounds__(256) void snn_recurrence(
    const float* __restrict__ cur,  // [schunk*T, H]
    const float* __restrict__ V1,   // [H, H]
    const float* __restrict__ bv1,  // [H]
    const float* __restrict__ W2,   // [1, H]
    const float* __restrict__ b2v,  // [1]
    const float* __restrict__ V2,   // [1, 1]
    const float* __restrict__ bv2,  // [1]
    float* __restrict__ out,        // [2*S]: spk2 then mem2
    float* __restrict__ state,      // [2*H + 2]
    int chunk, int schunk)
{
  extern __shared__ float lds[];
  float*    Vp   = lds;                            // H_*(H_+1)
  float*    w2s  = Vp + H_ * (H_ + 1);             // H_
  unsigned* mask = (unsigned*)(w2s + H_);          // 2*8 words

  const int j  = threadIdx.x;
  const int wv = j >> 5;
  const int ln = j & 31;

  // Stage V1 (padded rows: stride 257 -> bank (j+i)%64) and W2. Coalesced.
  for (int idx = j; idx < H_ * H_; idx += H_) {
    const int r = idx >> 8, c = idx & (H_ - 1);
    Vp[r * (H_ + 1) + c] = V1[idx];
  }
  w2s[j] = W2[j];

  float mem1, spkf;
  float mem2 = 0.f, spk2 = 0.f;
  if (chunk == 0) {
    mem1 = 0.f; spkf = 0.f;
  } else {
    mem1 = state[j];
    spkf = state[H_ + j];
    if (j == 0) { mem2 = state[2 * H_]; spk2 = state[2 * H_ + 1]; }
  }
  const float bv1j = bv1[j];
  const float b2s  = b2v[0];
  const float v2s  = V2[0];
  const float bv2s = bv2[0];

  // Seed the "previous spikes" mask into buffer 1 (iteration 0 reads it).
  {
    const unsigned bal = __builtin_amdgcn_ballot_w32(spkf > 0.f);
    if (ln == 0) mask[8 + wv] = bal;
  }
  __syncthreads();

  const int nsteps = schunk * T_;
  for (int n = 0; n < nsteps; ++n) {
    const int p = n & 1;                       // write buffer this step
    const unsigned* rd = mask + 8 * (p ^ 1);   // previous step's spikes

    const float c1 = cur[(size_t)n * H_ + j];
    if (n + 1 < nsteps)  // global_prefetch_b8 hint for next step's currents
      __builtin_prefetch(&cur[(size_t)(n + 1) * H_ + j], 0, 1);

    // Recurrent term: sum V1[j, i] over set bits of the previous spike mask.
    float rec = 0.f;
#pragma unroll
    for (int w = 0; w < 8; ++w) {
      unsigned bits = rd[w];
      const float* vrow = Vp + j * (H_ + 1) + w * 32;
      while (bits) {
        rec += vrow[__builtin_ctz(bits)];
        bits &= bits - 1;
      }
    }

    // LIF update (subtract reset, from PREVIOUS mem — snntorch RLeaky).
    const float reset = (mem1 > THRV) ? THRV : 0.f;
    mem1 = BETA * mem1 + c1 + rec + bv1j - reset;
    const bool spk = (mem1 - THRV) > 0.f;

    // Publish new spike mask (wave32 ballot, one word per wave).
    const unsigned bal = __builtin_amdgcn_ballot_w32(spk);
    if (ln == 0) mask[8 * p + wv] = bal;
    __syncthreads();   // the ONLY barrier per step

    if (j == 0) {
      const unsigned* nw = mask + 8 * p;       // NEW spikes drive neuron 2
      float cur2 = b2s;
#pragma unroll
      for (int w = 0; w < 8; ++w) {
        unsigned bits = nw[w];
        const float* wrow = w2s + w * 32;
        while (bits) {
          cur2 += wrow[__builtin_ctz(bits)];
          bits &= bits - 1;
        }
      }
      const float reset2 = (mem2 > THRV) ? THRV : 0.f;
      mem2 = BETA * mem2 + cur2 + spk2 * v2s + bv2s - reset2;
      spk2 = ((mem2 - THRV) > 0.f) ? 1.f : 0.f;
      if ((n & (T_ - 1)) == (T_ - 1)) {        // end of a sample
        const int sg = chunk * schunk + (n >> 7);
        out[sg]      = spk2;
        out[S_ + sg] = mem2;
      }
    }
    spkf = spk ? 1.f : 0.f;
  }

  // Persist carry for next chunk.
  state[j]      = mem1;
  state[H_ + j] = spkf;
  if (j == 0) { state[2 * H_] = mem2; state[2 * H_ + 1] = spk2; }
}

// ---------------------------------------------------------------------------
// Host launcher. Inputs (setup_inputs order):
//   0:x 1:W1 2:b1 3:V1 4:bv1 5:W2 6:b2 7:V2 8:bv2
// Workspace: [0,4096) carry state; rest = cur buffer (chunk of samples).
// ---------------------------------------------------------------------------
extern "C" void kernel_launch(void* const* d_in, const int* in_sizes, int n_in,
                              void* d_out, int out_size, void* d_ws, size_t ws_size,
                              hipStream_t stream)
{
  (void)in_sizes; (void)n_in; (void)out_size;
  const float* x   = (const float*)d_in[0];
  const float* W1  = (const float*)d_in[1];
  const float* b1  = (const float*)d_in[2];
  const float* V1  = (const float*)d_in[3];
  const float* bv1 = (const float*)d_in[4];
  const float* W2  = (const float*)d_in[5];
  const float* b2  = (const float*)d_in[6];
  const float* V2  = (const float*)d_in[7];
  const float* bv2 = (const float*)d_in[8];
  float* out = (float*)d_out;

  float* state = (float*)d_ws;
  float* cur   = (float*)((char*)d_ws + 4096);
  const size_t cap = (ws_size > 4096) ? ws_size - 4096 : 0;

  // Largest power-of-two sample chunk whose cur buffer fits the workspace.
  int schunk = S_;
  while (schunk > 1 &&
         (size_t)schunk * T_ * H_ * sizeof(float) > cap)
    schunk >>= 1;

  const int nchunks = S_ / schunk;
  const size_t ldsBytes =
      (size_t)(H_ * (H_ + 1) + H_) * sizeof(float)   // Vp + w2s
      + (size_t)(16 + 8) * sizeof(unsigned);         // mask (+pad)

  for (int ch = 0; ch < nchunks; ++ch) {
    dim3 grid(schunk * T_ / 16);
    snn_gemm_cur<<<grid, 256, 0, stream>>>(x, W1, b1, cur, ch, schunk);
    snn_recurrence<<<1, 256, ldsBytes, stream>>>(cur, V1, bv1, W2, b2, V2, bv2,
                                                 out, state, ch, schunk);
  }
}
